// CausalFusionModel_45904610460056
// MI455X (gfx1250) — compile-verified, matrix operand order checked by reading, not run on hardware
//
#include <hip/hip_runtime.h>
#include <cstddef>

// ---------------------------------------------------------------- constants
#define DIMC   1536
#define HEADSC 12
#define HDC    128
#define FFNC   6144
#define NVID   3120   // 3 * 1040
#define NAUD   960    // 15 * 64
#define FVID   3
#define LVID   1040
#define FAUD   15
#define LAUD   64
#define CTXN   512
#define EPSF   1e-6f

typedef __bf16 bf16;
typedef __attribute__((ext_vector_type(16))) __bf16 v16bf;
typedef __attribute__((ext_vector_type(8)))  float  v8f;
typedef int v4i __attribute__((ext_vector_type(4)));

#if defined(__HIP_DEVICE_COMPILE__) && \
    __has_builtin(__builtin_amdgcn_global_load_async_to_lds_b128)
#define ASYNC_LDS 1
// builtin expects int4* operands (per hipcc diagnostic); qualify with the
// proper address spaces (generic is a superset, so this converts either way)
typedef __attribute__((address_space(1))) v4i* gas_v4i_p;
typedef __attribute__((address_space(3))) v4i* las_v4i_p;
#else
#define ASYNC_LDS 0
#endif

__device__ __forceinline__ void wait_async0() {
#if __has_builtin(__builtin_amdgcn_s_wait_asynccnt)
  __builtin_amdgcn_s_wait_asynccnt(0);
#else
  asm volatile("s_wait_asynccnt 0" ::: "memory");
#endif
}

// K-index mapping for 16-bit A/B WMMA fragments (ISA 7.12.2):
// lanes 0-15 hold K = {0..7, 16..23}, lanes 16-31 hold K = {8..15, 24..31}
__device__ __forceinline__ int kmap(int e, int h) {
  return (e < 8) ? (h * 8 + e) : (16 + h * 8 + (e - 8));
}

__device__ __forceinline__ float gelu_tanh(float x) {
  float x3 = x * x * x;
  return 0.5f * x * (1.f + tanhf(0.7978845608028654f * (x + 0.044715f * x3)));
}

// ---------------------------------------------------------------- GEMM
// C[M,N] = A[M,K](bf16) @ W[K,N](f32 -> bf16 on the fly) + bias
// Block: 16 (M) x 128 (N); 8 waves, one 16x16 WMMA tile each along N.
// A tile staged with async global->LDS DMA (ASYNCcnt) by wave 0;
// W tile staged synchronously (needs f32->bf16 conversion in flight).
template <int OUT_BF16, int DO_GELU>
__global__ __launch_bounds__(256) void k_gemm(
    const bf16* __restrict__ A, const float* __restrict__ W,
    const float* __restrict__ bias, void* __restrict__ Cout,
    int M, int N, int K) {
  __shared__ __align__(16) bf16 As[16][48];   // 16x32 tile (+16B-aligned pad)
  __shared__ bf16 Ws[32][136];                // 32x128 tile (+pad)
  const int m0   = blockIdx.y * 16;
  const int n0   = blockIdx.x * 128;
  const int tid  = threadIdx.x;
  const int lane = tid & 31;
  const int wv   = tid >> 5;     // 0..7
  const int l15  = lane & 15;
  const int h    = lane >> 4;

  v8f c;
#pragma unroll
  for (int r = 0; r < 8; ++r) c[r] = 0.f;

  for (int k0 = 0; k0 < K; k0 += 32) {
#if ASYNC_LDS
    // A tile: 64 x 16B chunks; wave 0 issues 2 async b128 copies per lane
    if (wv == 0) {
#pragma unroll
      for (int i = 0; i < 2; ++i) {
        int idx = i * 32 + lane;
        int r = idx >> 2, cb = (idx & 3) * 8;  // 8 bf16 = 16 bytes
        __builtin_amdgcn_global_load_async_to_lds_b128(
            (gas_v4i_p)&A[(size_t)(m0 + r) * K + k0 + cb],
            (las_v4i_p)&As[r][cb], 0, 0);
      }
    }
#else
#pragma unroll
    for (int i = 0; i < 2; ++i) {
      int idx = tid * 2 + i;
      int r = idx >> 5, cc = idx & 31;
      As[r][cc] = A[(size_t)(m0 + r) * K + k0 + cc];
    }
#endif
    // stage W tile with f32->bf16 conversion
#pragma unroll
    for (int i = 0; i < 16; ++i) {
      int idx = tid + i * 256;
      int r = idx >> 7, cc = idx & 127;
      Ws[r][cc] = (bf16)W[(size_t)(k0 + r) * N + n0 + cc];
    }
    if (k0 + 32 < K)  // hint next W tile into cache (global_prefetch)
      __builtin_prefetch(&W[(size_t)(k0 + 32) * N + n0 + (tid & 127)], 0, 1);
#if ASYNC_LDS
    if (wv == 0) wait_async0();  // drain ASYNCcnt before the block barrier
#endif
    __syncthreads();

    v16bf a, b;
#pragma unroll
    for (int e = 0; e < 16; ++e) a[e] = As[l15][kmap(e, h)];
#pragma unroll
    for (int e = 0; e < 16; ++e) b[e] = Ws[kmap(e, h)][wv * 16 + l15];
    c = __builtin_amdgcn_wmma_f32_16x16x32_bf16(false, a, false, b,
                                                (short)0, c, false, false);
    __syncthreads();
  }

#pragma unroll
  for (int r = 0; r < 8; ++r) {
    int row = m0 + h * 8 + r;
    int col = n0 + wv * 16 + l15;
    float v = c[r] + (bias ? bias[col] : 0.f);
    if (DO_GELU) v = gelu_tanh(v);
    if (OUT_BF16)
      ((bf16*)Cout)[(size_t)row * N + col] = (bf16)v;
    else
      ((float*)Cout)[(size_t)row * N + col] = v;
  }
}

// ---------------------------------------------------------------- attention
// One wave handles one (16-query tile, head). Flash online softmax.
// Lfrm > 0: block-causal by frame of length Lfrm; Lfrm == 0: dense.
__global__ __launch_bounds__(256) void k_attn(
    const bf16* __restrict__ Q, const bf16* __restrict__ Km,
    const bf16* __restrict__ V, bf16* __restrict__ O,
    int Nq, int Nk, int Lfrm, float scale) {
  __shared__ float Plds[8][16][32];
  const int wv   = threadIdx.x >> 5;
  const int lane = threadIdx.x & 31;
  const int tiles = (Nq / 16) * HEADSC;
  const int tile  = blockIdx.x * 8 + wv;
  if (tile >= tiles) return;  // whole-wave exit; no block-wide barriers used
  const int head = tile % HEADSC;
  const int qt   = tile / HEADSC;
  const int l15  = lane & 15;
  const int h    = lane >> 4;
  const int q0   = qt * 16;
  const int kmax = Lfrm ? ((q0 / Lfrm) + 1) * Lfrm : Nk;

  // Q fragments: 4 chunks of 16x32 over HD=128
  v16bf aq[4];
#pragma unroll
  for (int cch = 0; cch < 4; ++cch)
#pragma unroll
    for (int e = 0; e < 16; ++e)
      aq[cch][e] =
          Q[(size_t)(q0 + l15) * DIMC + head * HDC + cch * 32 + kmap(e, h)];

  float mi[8], li[8];
  v8f acc[8];
#pragma unroll
  for (int r = 0; r < 8; ++r) { mi[r] = -1e30f; li[r] = 0.f; }
#pragma unroll
  for (int cc = 0; cc < 8; ++cc)
#pragma unroll
    for (int r = 0; r < 8; ++r) acc[cc][r] = 0.f;

  for (int ks = 0; ks < kmax; ks += 32) {
    const bool v1 = (ks + 16 < kmax);
    v8f s0, s1;
#pragma unroll
    for (int r = 0; r < 8; ++r) { s0[r] = 0.f; s1[r] = 0.f; }

    // S = Q K^T : B fragment = K^T (32 hd-K x 16 key cols)
#pragma unroll
    for (int cch = 0; cch < 4; ++cch) {
      v16bf b;
#pragma unroll
      for (int e = 0; e < 16; ++e)
        b[e] = Km[(size_t)(ks + l15) * DIMC + head * HDC + cch * 32 +
                  kmap(e, h)];
      s0 = __builtin_amdgcn_wmma_f32_16x16x32_bf16(false, aq[cch], false, b,
                                                   (short)0, s0, false, false);
    }
    if (v1) {
#pragma unroll
      for (int cch = 0; cch < 4; ++cch) {
        v16bf b;
#pragma unroll
        for (int e = 0; e < 16; ++e)
          b[e] = Km[(size_t)(ks + 16 + l15) * DIMC + head * HDC + cch * 32 +
                    kmap(e, h)];
        s1 = __builtin_amdgcn_wmma_f32_16x16x32_bf16(
            false, aq[cch], false, b, (short)0, s1, false, false);
      }
    }
#pragma unroll
    for (int r = 0; r < 8; ++r) {
      s0[r] *= scale;
      s1[r] = v1 ? s1[r] * scale : -1e30f;
    }

    // online softmax over 32 columns (16-lane xor reductions per half)
    float mnew[8], al[8];
#pragma unroll
    for (int r = 0; r < 8; ++r) {
      float mx = fmaxf(s0[r], s1[r]);
#pragma unroll
      for (int o = 8; o >= 1; o >>= 1) mx = fmaxf(mx, __shfl_xor(mx, o, 32));
      mnew[r] = fmaxf(mi[r], mx);
    }
#pragma unroll
    for (int r = 0; r < 8; ++r) {
      s0[r] = __expf(s0[r] - mnew[r]);
      s1[r] = __expf(s1[r] - mnew[r]);
      float rs = s0[r] + s1[r];
#pragma unroll
      for (int o = 8; o >= 1; o >>= 1) rs += __shfl_xor(rs, o, 32);
      al[r] = __expf(mi[r] - mnew[r]);
      li[r] = li[r] * al[r] + rs;
      mi[r] = mnew[r];
    }
#pragma unroll
    for (int cc = 0; cc < 8; ++cc)
#pragma unroll
      for (int r = 0; r < 8; ++r) acc[cc][r] *= al[r];

    // reshuffle P (D-layout) -> A-layout via per-wave LDS
#pragma unroll
    for (int r = 0; r < 8; ++r) {
      Plds[wv][h * 8 + r][l15]      = s0[r];
      Plds[wv][h * 8 + r][16 + l15] = s1[r];
    }
    asm volatile("s_wait_dscnt 0" ::: "memory");
    v16bf ap;
#pragma unroll
    for (int e = 0; e < 16; ++e) ap[e] = (bf16)Plds[wv][l15][kmap(e, h)];

    // O += P V  (8 output chunks of 16 hd-cols)
#pragma unroll
    for (int cc = 0; cc < 8; ++cc) {
      v16bf bv;
#pragma unroll
      for (int e = 0; e < 16; ++e) {
        int kr = ks + kmap(e, h);
        bv[e] = (kr < Nk)
                    ? V[(size_t)kr * DIMC + head * HDC + cc * 16 + l15]
                    : (bf16)0.f;
      }
      acc[cc] = __builtin_amdgcn_wmma_f32_16x16x32_bf16(
          false, ap, false, bv, (short)0, acc[cc], false, false);
    }
  }

#pragma unroll
  for (int cc = 0; cc < 8; ++cc)
#pragma unroll
    for (int r = 0; r < 8; ++r) {
      int row = q0 + h * 8 + r;
      int col = head * HDC + cc * 16 + l15;
      O[(size_t)row * DIMC + col] = (bf16)(acc[cc][r] / li[r]);
    }
}

// ---------------------------------------------------------------- elementwise
__global__ void k_mod(float* __restrict__ out, const float* __restrict__ e,
                      const float* __restrict__ pmod, int tot) {
  int i = blockIdx.x * blockDim.x + threadIdx.x;
  if (i < tot) out[i] = e[i] + pmod[i % (6 * DIMC)];
}

// per-token LayerNorm, then y = xhat*(scale(+1)) + shift, bf16 out.
// frame_stride>0: scale/shift indexed [ (t/Ltok)*frame_stride + d ].
__global__ __launch_bounds__(256) void k_ln(
    const float* __restrict__ X, bf16* __restrict__ Y,
    const float* __restrict__ scale, const float* __restrict__ shift,
    int frame_stride, int Ltok, int add_one) {
  __shared__ float red[256];
  const int t = blockIdx.x;
  const float* x = X + (size_t)t * DIMC;
  float s = 0.f;
  for (int d = threadIdx.x; d < DIMC; d += 256) s += x[d];
  red[threadIdx.x] = s;
  __syncthreads();
  for (int o = 128; o > 0; o >>= 1) {
    if (threadIdx.x < o) red[threadIdx.x] += red[threadIdx.x + o];
    __syncthreads();
  }
  const float mean = red[0] / DIMC;
  __syncthreads();
  float v = 0.f;
  for (int d = threadIdx.x; d < DIMC; d += 256) {
    float e = x[d] - mean;
    v += e * e;
  }
  red[threadIdx.x] = v;
  __syncthreads();
  for (int o = 128; o > 0; o >>= 1) {
    if (threadIdx.x < o) red[threadIdx.x] += red[threadIdx.x + o];
    __syncthreads();
  }
  const float rstd = rsqrtf(red[0] / DIMC + EPSF);
  const int fo = frame_stride ? (t / Ltok) * frame_stride : 0;
  for (int d = threadIdx.x; d < DIMC; d += 256) {
    float xh = (x[d] - mean) * rstd;
    float sc = scale[fo + d] + (add_one ? 1.f : 0.f);
    Y[(size_t)t * DIMC + d] = (bf16)(xh * sc + shift[fo + d]);
  }
}

// RMS-norm over DIM with gain, optional per-head RoPE, bf16 pack.
__global__ __launch_bounds__(256) void k_rmsrope(
    const float* __restrict__ X, bf16* __restrict__ Y,
    const float* __restrict__ g, const float* __restrict__ freqs,
    int do_rope) {
  __shared__ float red[256];
  const int t = blockIdx.x;
  const float* x = X + (size_t)t * DIMC;
  float s = 0.f;
  for (int d = threadIdx.x; d < DIMC; d += 256) s += x[d] * x[d];
  red[threadIdx.x] = s;
  __syncthreads();
  for (int o = 128; o > 0; o >>= 1) {
    if (threadIdx.x < o) red[threadIdx.x] += red[threadIdx.x + o];
    __syncthreads();
  }
  const float r = rsqrtf(red[0] / DIMC + EPSF);
  for (int p = threadIdx.x; p < DIMC / 2; p += 256) {
    int d0 = 2 * p, d1 = 2 * p + 1;
    float a = x[d0] * r * g[d0];
    float b = x[d1] * r * g[d1];
    if (do_rope) {
      float fr = freqs[(size_t)t * (HDC / 2) + (p & 63)];
      float cs = __cosf(fr), sn = __sinf(fr);
      float na = a * cs - b * sn;
      float nb = a * sn + b * cs;
      a = na; b = nb;
    }
    Y[(size_t)t * DIMC + d0] = (bf16)a;
    Y[(size_t)t * DIMC + d1] = (bf16)b;
  }
}

// x += y * gate   (gate indexed per frame; gate==nullptr -> 1)
__global__ void k_resid(float* __restrict__ X, const float* __restrict__ Y,
                        const float* __restrict__ gate, int frame_stride,
                        int Ltok, int tot) {
  int i = blockIdx.x * blockDim.x + threadIdx.x;
  if (i >= tot) return;
  int t = i / DIMC, d = i % DIMC;
  float gt = gate ? gate[(size_t)(t / Ltok) * frame_stride + d] : 1.f;
  X[i] += Y[i] * gt;
}

__global__ void k_addbf(bf16* __restrict__ Z, const bf16* __restrict__ A,
                        const bf16* __restrict__ B, int n) {
  int i = blockIdx.x * blockDim.x + threadIdx.x;
  if (i < n) Z[i] = (bf16)((float)A[i] + (float)B[i]);
}

__global__ void k_cvt(bf16* __restrict__ Z, const float* __restrict__ A,
                      int n) {
  int i = blockIdx.x * blockDim.x + threadIdx.x;
  if (i < n) Z[i] = (bf16)A[i];
}

// ---------------------------------------------------------------- host side
struct Lin { const float* w; const float* b; };
struct Branch {
  const float *ca_gk, *ca_gkf, *ca_gq;
  Lin ca_k, ca_kf, ca_o, ca_q, ca_v, ca_vf, ffn1, ffn2;
  const float* mod;
  const float *norm3_b, *norm3_g, *pre_b, *pre_g;
  const float *sa_gk, *sa_gq;
  Lin sa_k, sa_o, sa_q, sa_v;
};

// jax pytree flattening order: sorted dict keys, {'b','w'} within each linear
static int fill_branch(void* const* din, int i, Branch& B) {
  auto F = [&](const float** p) { *p = (const float*)din[i++]; };
  auto L = [&](Lin* l) { F(&l->b); F(&l->w); };
  F(&B.ca_gk); F(&B.ca_gkf); F(&B.ca_gq);
  L(&B.ca_k); L(&B.ca_kf); L(&B.ca_o); L(&B.ca_q); L(&B.ca_v); L(&B.ca_vf);
  L(&B.ffn1); L(&B.ffn2);
  F(&B.mod);
  F(&B.norm3_b); F(&B.norm3_g); F(&B.pre_b); F(&B.pre_g);
  F(&B.sa_gk); F(&B.sa_gq);
  L(&B.sa_k); L(&B.sa_o); L(&B.sa_q); L(&B.sa_v);
  return i;
}

struct Scratch {
  float *xv, *xa, *og, *modv, *moda, *tmp32;
  bf16 *xin, *qb, *qrb, *kb, *vb, *yb, *tb, *ktb, *vtb, *x1b, *x2b, *ctxb,
      *ffn_h;
};

template <int OB, int OG>
static void gemm(const bf16* A, const float* W, const float* bias, void* C,
                 int M, int N, int K, hipStream_t st) {
  dim3 g(N / 128, M / 16);
  k_gemm<OB, OG><<<g, 256, 0, st>>>(A, W, bias, C, M, N, K);
}

static void attn(const bf16* Q, const bf16* K, const bf16* V, bf16* O, int Nq,
                 int Nk, int Lfrm, hipStream_t st) {
  int tiles = (Nq / 16) * HEADSC;
  k_attn<<<(tiles + 7) / 8, 256, 0, st>>>(Q, K, V, O, Nq, Nk, Lfrm,
                                          0.08838834764831845f);
}

static void run_sa(const Branch& P, float* x, float* modbuf, const float* e_in,
                   const float* freqs, int N, int F, int Ltok, Scratch& S,
                   hipStream_t st) {
  int tot = F * 6 * DIMC;
  k_mod<<<(tot + 255) / 256, 256, 0, st>>>(modbuf, e_in, P.mod, tot);
  k_ln<<<N, 256, 0, st>>>(x, S.xin, modbuf + 1 * DIMC, modbuf + 0 * DIMC,
                          6 * DIMC, Ltok, 1);
  gemm<0, 0>(S.xin, P.sa_q.w, P.sa_q.b, S.tmp32, N, DIMC, DIMC, st);
  k_rmsrope<<<N, 256, 0, st>>>(S.tmp32, S.qb, P.sa_gq, freqs, 1);
  gemm<0, 0>(S.xin, P.sa_k.w, P.sa_k.b, S.tmp32, N, DIMC, DIMC, st);
  k_rmsrope<<<N, 256, 0, st>>>(S.tmp32, S.kb, P.sa_gk, freqs, 1);
  gemm<1, 0>(S.xin, P.sa_v.w, P.sa_v.b, S.vb, N, DIMC, DIMC, st);
  attn(S.qb, S.kb, S.vb, S.yb, N, N, Ltok, st);  // block-causal by frame
  gemm<0, 0>(S.yb, P.sa_o.w, P.sa_o.b, S.tmp32, N, DIMC, DIMC, st);
  k_resid<<<(N * DIMC + 255) / 256, 256, 0, st>>>(x, S.tmp32,
                                                  modbuf + 2 * DIMC, 6 * DIMC,
                                                  Ltok, N * DIMC);
}

static void run_fusion(const Branch& P, float* x, const float* modbuf,
                       const float* freqs, int N, int Ltok,
                       const float* target, const float* tfreqs, int Nt,
                       const float* ctx, Scratch& S, hipStream_t st) {
  // h = LN(x)*norm3_g + norm3_b ; q = rms(h @ Wq)
  k_ln<<<N, 256, 0, st>>>(x, S.xin, P.norm3_g, P.norm3_b, 0, 1, 0);
  gemm<0, 0>(S.xin, P.ca_q.w, P.ca_q.b, S.tmp32, N, DIMC, DIMC, st);
  k_rmsrope<<<N, 256, 0, st>>>(S.tmp32, S.qb, P.ca_gq, nullptr, 0);
  k_rmsrope<<<N, 256, 0, st>>>(S.tmp32, S.qrb, P.ca_gq, freqs, 1);
  // context K/V
  k_cvt<<<(CTXN * DIMC + 255) / 256, 256, 0, st>>>(S.ctxb, ctx, CTXN * DIMC);
  gemm<0, 0>(S.ctxb, P.ca_k.w, P.ca_k.b, S.tmp32, CTXN, DIMC, DIMC, st);
  k_rmsrope<<<CTXN, 256, 0, st>>>(S.tmp32, S.kb, P.ca_gk, nullptr, 0);
  gemm<1, 0>(S.ctxb, P.ca_v.w, P.ca_v.b, S.vb, CTXN, DIMC, DIMC, st);
  attn(S.qb, S.kb, S.vb, S.x1b, N, CTXN, 0, st);
  // target K/V (t = LN(target)*pre_g + pre_b)
  k_ln<<<Nt, 256, 0, st>>>(target, S.tb, P.pre_g, P.pre_b, 0, 1, 0);
  gemm<0, 0>(S.tb, P.ca_kf.w, P.ca_kf.b, S.tmp32, Nt, DIMC, DIMC, st);
  k_rmsrope<<<Nt, 256, 0, st>>>(S.tmp32, S.ktb, P.ca_gkf, tfreqs, 1);
  gemm<1, 0>(S.tb, P.ca_vf.w, P.ca_vf.b, S.vtb, Nt, DIMC, DIMC, st);
  attn(S.qrb, S.ktb, S.vtb, S.x2b, N, Nt, 0, st);
  // x += (x1 + x2) @ Wo + bo
  k_addbf<<<(N * DIMC + 255) / 256, 256, 0, st>>>(S.yb, S.x1b, S.x2b,
                                                  N * DIMC);
  gemm<0, 0>(S.yb, P.ca_o.w, P.ca_o.b, S.tmp32, N, DIMC, DIMC, st);
  k_resid<<<(N * DIMC + 255) / 256, 256, 0, st>>>(x, S.tmp32, nullptr, 0, 1,
                                                  N * DIMC);
  // FFN with modulation rows e3/e4/e5
  k_ln<<<N, 256, 0, st>>>(x, S.xin, modbuf + 4 * DIMC, modbuf + 3 * DIMC,
                          6 * DIMC, Ltok, 1);
  gemm<1, 1>(S.xin, P.ffn1.w, P.ffn1.b, S.ffn_h, N, FFNC, DIMC, st);
  gemm<0, 0>(S.ffn_h, P.ffn2.w, P.ffn2.b, S.tmp32, N, DIMC, FFNC, st);
  k_resid<<<(N * DIMC + 255) / 256, 256, 0, st>>>(x, S.tmp32,
                                                  modbuf + 5 * DIMC, 6 * DIMC,
                                                  Ltok, N * DIMC);
}

extern "C" void kernel_launch(void* const* d_in, const int* in_sizes, int n_in,
                              void* d_out, int out_size, void* d_ws,
                              size_t ws_size, hipStream_t stream) {
  (void)in_sizes; (void)out_size; (void)ws_size;
  if (n_in < 76) return;

  const float* vid      = (const float*)d_in[0];
  const float* audio    = (const float*)d_in[1];
  const float* vid_e    = (const float*)d_in[2];
  const float* audio_e  = (const float*)d_in[3];
  const float* vfreqs   = (const float*)d_in[4];
  const float* afreqs   = (const float*)d_in[5];
  const float* vctx     = (const float*)d_in[6];
  const float* actx     = (const float*)d_in[7];

  Branch PA, PV;  // params dict: 'audio' sorts before 'vid'
  int pi = 8;
  pi = fill_branch(d_in, pi, PA);
  pi = fill_branch(d_in, pi, PV);

  // -------- workspace carve (256B aligned) --------
  char* base = (char*)d_ws;
  size_t off = 0;
  auto carve = [&](size_t bytes) {
    void* p = base + off;
    off = (off + bytes + 255) & ~(size_t)255;
    return p;
  };
  Scratch S;
  S.xv    = (float*)carve((size_t)NVID * DIMC * 4);
  S.xa    = (float*)carve((size_t)NAUD * DIMC * 4);
  S.og    = (float*)carve((size_t)NAUD * DIMC * 4);
  S.modv  = (float*)carve((size_t)FVID * 6 * DIMC * 4);
  S.moda  = (float*)carve((size_t)FAUD * 6 * DIMC * 4);
  S.tmp32 = (float*)carve((size_t)NVID * DIMC * 4);
  S.xin   = (bf16*)carve((size_t)NVID * DIMC * 2);
  S.qb    = (bf16*)carve((size_t)NVID * DIMC * 2);
  S.qrb   = (bf16*)carve((size_t)NVID * DIMC * 2);
  S.kb    = (bf16*)carve((size_t)NVID * DIMC * 2);
  S.vb    = (bf16*)carve((size_t)NVID * DIMC * 2);
  S.yb    = (bf16*)carve((size_t)NVID * DIMC * 2);
  S.tb    = (bf16*)carve((size_t)NVID * DIMC * 2);
  S.ktb   = (bf16*)carve((size_t)NVID * DIMC * 2);
  S.vtb   = (bf16*)carve((size_t)NVID * DIMC * 2);
  S.x1b   = (bf16*)carve((size_t)NVID * DIMC * 2);
  S.x2b   = (bf16*)carve((size_t)NVID * DIMC * 2);
  S.ctxb  = (bf16*)carve((size_t)CTXN * DIMC * 2);
  S.ffn_h = (bf16*)carve((size_t)NVID * FFNC * 2);

  // residual working copies (inputs stay untouched)
  (void)hipMemcpyAsync(S.xv, vid, (size_t)NVID * DIMC * 4,
                       hipMemcpyDeviceToDevice, stream);
  (void)hipMemcpyAsync(S.xa, audio, (size_t)NAUD * DIMC * 4,
                       hipMemcpyDeviceToDevice, stream);

  // self-attention branches
  run_sa(PA, S.xa, S.moda, audio_e, afreqs, NAUD, FAUD, LAUD, S, stream);
  run_sa(PV, S.xv, S.modv, vid_e, vfreqs, NVID, FVID, LVID, S, stream);

  // og_audio = audio2 (needed by vid fusion after audio fusion runs)
  (void)hipMemcpyAsync(S.og, S.xa, (size_t)NAUD * DIMC * 4,
                       hipMemcpyDeviceToDevice, stream);

  // audio3 = fusion(audio2, target=vid2); vid3 = fusion(vid2, target=og_audio)
  run_fusion(PA, S.xa, S.moda, afreqs, NAUD, LAUD, S.xv, vfreqs, NVID, actx,
             S, stream);
  run_fusion(PV, S.xv, S.modv, vfreqs, NVID, LVID, S.og, afreqs, NAUD, vctx,
             S, stream);

  // outputs: (vid3, audio3) concatenated, fp32
  (void)hipMemcpyAsync(d_out, S.xv, (size_t)NVID * DIMC * 4,
                       hipMemcpyDeviceToDevice, stream);
  (void)hipMemcpyAsync((char*)d_out + (size_t)NVID * DIMC * 4, S.xa,
                       (size_t)NAUD * DIMC * 4, hipMemcpyDeviceToDevice,
                       stream);
}
